// SlidingWindowAttention_6528350290244
// MI455X (gfx1250) — matrix operand order repaired and checked by reading, not compiled, not run on hardware
//
#include <hip/hip_runtime.h>

#define DIMN      1024
#define NHEADS    16
#define HDIM      64
#define WINDOW    512
#define SEQ       4096

typedef __attribute__((ext_vector_type(16))) __bf16 v16bf;
typedef __attribute__((ext_vector_type(8)))  __bf16 v8bf;
typedef __attribute__((ext_vector_type(8)))  float  v8f;

// ---- fragment loaders (contiguous k = 16*(lane/16)+s, same in A and B) ----------------------

// B 32x16 (16-bit): lane holds 16 contiguous K at offset 16*(lane/16)
static __device__ __forceinline__ v16bf ld_b_frag(const __bf16* p) {
    return *(const v16bf*)p;
}

// A 16x32 (16-bit): lane holds 16 contiguous K at offset 16*hl
static __device__ __forceinline__ v16bf ld_a_frag(const __bf16* p, int hl) {
    return *(const v16bf*)(p + 16 * hl);
}

static __device__ __forceinline__ v8f wmma_bf16(v16bf a, v16bf b, v8f c) {
    return __builtin_amdgcn_wmma_f32_16x16x32_bf16(
        /*neg_a=*/false, a, /*neg_b=*/false, b,
        /*c_mod=*/(short)0, c, /*reuse_a=*/false, /*reuse_b=*/false);
}

// ---- DPP16 16-lane butterfly reductions (no LDS traffic) -------------------
// quad_perm xor1/xor2 make quads uniform; then row_half_mirror (xor7 == xor4
// on quad-uniform data) and row_mirror (xor15 == xor8 on octet-uniform data)
// complete the 16-lane reduction. Rows of 16 never cross the wave32 halves.

template <int CTRL>
static __device__ __forceinline__ float dppf(float v) {
    return __int_as_float(__builtin_amdgcn_mov_dpp(
        __float_as_int(v), CTRL, /*row_mask=*/0xf, /*bank_mask=*/0xf,
        /*bound_ctrl=*/true));
}
static __device__ __forceinline__ float red_max16(float v) {
    v = fmaxf(v, dppf<0xB1>(v));    // quad_perm [1,0,3,2] : xor 1
    v = fmaxf(v, dppf<0x4E>(v));    // quad_perm [2,3,0,1] : xor 2
    v = fmaxf(v, dppf<0x141>(v));   // row_half_mirror     : xor 4
    v = fmaxf(v, dppf<0x140>(v));   // row_mirror          : xor 8
    return v;
}
static __device__ __forceinline__ float red_sum16(float v) {
    v += dppf<0xB1>(v);
    v += dppf<0x4E>(v);
    v += dppf<0x141>(v);
    v += dppf<0x140>(v);
    return v;
}

// ---- fp32 -> bf16 conversion ----------------------------------------------

__global__ void f32_to_bf16_kernel(const float* __restrict__ in,
                                   __bf16* __restrict__ out, int n) {
    int i = blockIdx.x * blockDim.x + threadIdx.x;
    if (i < n) out[i] = (__bf16)in[i];
}

// ---- WMMA GEMM: C = A[M,K] * W[N,K]^T, 32x64 tile per wave -----------------
// 2 M-fragments x 4 N-fragments -> 8 WMMAs per k-step on 12 b128 loads
// (~21 FLOP/B from L2), keeping the matrix pipes saturated.
// MODE 0: bf16 row-major [M,N]   MODE 1: bf16 transposed [N,M]   MODE 2: f32 [M,N]
template <int MODE>
__global__ void __launch_bounds__(256)
gemm_bf16_wmma_kernel(const __bf16* __restrict__ A,
                      const __bf16* __restrict__ W,
                      void* __restrict__ out, int M, int N, int K) {
    const int lane = threadIdx.x & 31;
    const int wave = threadIdx.x >> 5;
    const int lm = lane & 15;
    const int hl = lane >> 4;
    const int ntile = N >> 6;                                  // 64-wide N tiles
    const int tile = blockIdx.x * (blockDim.x >> 5) + wave;
    const int m0 = (tile / ntile) << 5;                        // 32 rows
    const int n0 = (tile % ntile) << 6;                        // 64 cols

    const __bf16* arow0 = A + (size_t)(m0 + lm) * K;
    const __bf16* arow1 = arow0 + (size_t)16 * K;
    const __bf16* wrow0 = W + (size_t)(n0 + lm) * K;
    const __bf16* wrow1 = wrow0 + (size_t)16 * K;
    const __bf16* wrow2 = wrow0 + (size_t)32 * K;
    const __bf16* wrow3 = wrow0 + (size_t)48 * K;

    v8f acc[2][4] = {};
    for (int k = 0; k < K; k += 32) {
        __builtin_prefetch(arow0 + k + 256, 0, 1);   // global_prefetch_b8
        __builtin_prefetch(wrow0 + k + 256, 0, 1);
        const v16bf a0 = ld_a_frag(arow0 + k, hl);
        const v16bf a1 = ld_a_frag(arow1 + k, hl);
        const v16bf b0 = ld_b_frag(wrow0 + k + 16 * hl);
        const v16bf b1 = ld_b_frag(wrow1 + k + 16 * hl);
        const v16bf b2 = ld_b_frag(wrow2 + k + 16 * hl);
        const v16bf b3 = ld_b_frag(wrow3 + k + 16 * hl);
        acc[0][0] = wmma_bf16(a0, b0, acc[0][0]);
        acc[0][1] = wmma_bf16(a0, b1, acc[0][1]);
        acc[0][2] = wmma_bf16(a0, b2, acc[0][2]);
        acc[0][3] = wmma_bf16(a0, b3, acc[0][3]);
        acc[1][0] = wmma_bf16(a1, b0, acc[1][0]);
        acc[1][1] = wmma_bf16(a1, b1, acc[1][1]);
        acc[1][2] = wmma_bf16(a1, b2, acc[1][2]);
        acc[1][3] = wmma_bf16(a1, b3, acc[1][3]);
    }
#pragma unroll
    for (int mi = 0; mi < 2; ++mi) {
#pragma unroll
        for (int ji = 0; ji < 4; ++ji) {
#pragma unroll
            for (int r = 0; r < 8; ++r) {
                const int row = m0 + mi * 16 + r + 8 * hl;  // C: M = r + 8*(lane/16)
                const int col = n0 + ji * 16 + lm;          //    N = lane%16
                if (MODE == 0)      ((__bf16*)out)[(size_t)row * N + col] = (__bf16)acc[mi][ji][r];
                else if (MODE == 1) ((__bf16*)out)[(size_t)col * M + row] = (__bf16)acc[mi][ji][r];
                else                ((float*)out)[(size_t)row * N + col] = acc[mi][ji][r];
            }
        }
    }
}

// ---- sliding-window flash attention ---------------------------------------
// One wave per (head, 16-query tile). Q,K: [SEQ,DIMN] bf16; Vt: [DIMN,SEQ] bf16.
// Softmax runs in the exp2 domain: SCALE = 1/sqrt(64) * log2(e), exponentials
// are bare v_exp_f32 trans-ops that co-execute with the WMMA pipe.
#define SSCALE 0.18033688011112042f   // 0.125 * log2(e)

__global__ void __launch_bounds__(256)
swa_flash_wmma_kernel(const __bf16* __restrict__ Q,
                      const __bf16* __restrict__ Kb,
                      const __bf16* __restrict__ Vt,
                      __bf16* __restrict__ O) {
    __shared__ __bf16 pbuf[8][16 * 32];   // per-wave P staging (C-layout -> A-layout)

    const int lane = threadIdx.x & 31;
    const int wave = threadIdx.x >> 5;
    const int lm = lane & 15;
    const int hl = lane >> 4;
    const int gid = blockIdx.x * 8 + wave;
    const int h  = gid >> 8;               // 256 q-tiles per head
    const int q0 = (gid & 255) << 4;
    const int hb = h * HDIM;
    __bf16* pb = &pbuf[wave][0];

    // Q A-fragments: head_dim 64 = two K-chunks of 32
    const __bf16* qrow = Q + (size_t)(q0 + lm) * DIMN + hb;
    const v16bf qa0 = ld_a_frag(qrow, hl);
    const v16bf qa1 = ld_a_frag(qrow + 32, hl);

    v8f acc0 = {}, acc1 = {}, acc2 = {}, acc3 = {};
    float m[8], l[8];
#pragma unroll
    for (int r = 0; r < 8; ++r) { m[r] = -1e30f; l[r] = 0.f; }

    int s_lo = q0 - (WINDOW - 1);
    if (s_lo < 0) s_lo = 0;
    s_lo &= ~31;                           // 32-aligned; mask handles the edges
    const int s_hi = q0 + 15;

    for (int s0 = s_lo; s0 <= s_hi; s0 += 32) {
        // --- S = Q K^T for keys [s0, s0+32) : two 16x16 score tiles ---
        const __bf16* k0row = Kb + (size_t)(s0 + lm) * DIMN + hb;
        const __bf16* k1row = k0row + (size_t)16 * DIMN;
        v8f st0 = {}, st1 = {};
        st0 = wmma_bf16(qa0, ld_b_frag(k0row + 16 * hl), st0);
        st0 = wmma_bf16(qa1, ld_b_frag(k0row + 32 + 16 * hl), st0);
        st1 = wmma_bf16(qa0, ld_b_frag(k1row + 16 * hl), st1);
        st1 = wmma_bf16(qa1, ld_b_frag(k1row + 32 + 16 * hl), st1);

        // --- online softmax (row = r + 8*hl lives across 16 lanes of one half) ---
#pragma unroll
        for (int r = 0; r < 8; ++r) {
            const int qi   = q0 + r + 8 * hl;
            const int rel0 = qi - (s0 + lm);
            const int rel1 = rel0 - 16;
            const bool ok0 = (rel0 >= 0) && (rel0 < WINDOW);
            const bool ok1 = (rel1 >= 0) && (rel1 < WINDOW);
            const float v0 = ok0 ? st0[r] * SSCALE : -1e30f;
            const float v1 = ok1 ? st1[r] * SSCALE : -1e30f;
            const float t = red_max16(fmaxf(v0, v1));
            const float mnew = fmaxf(m[r], t);
            const float resc = __builtin_amdgcn_exp2f(m[r] - mnew);
            m[r] = mnew;
            const float e0 = ok0 ? __builtin_amdgcn_exp2f(v0 - mnew) : 0.f;
            const float e1 = ok1 ? __builtin_amdgcn_exp2f(v1 - mnew) : 0.f;
            const float ps = red_sum16(e0 + e1);
            l[r] = l[r] * resc + ps;
            acc0[r] *= resc; acc1[r] *= resc; acc2[r] *= resc; acc3[r] *= resc;
            const int prow = r + 8 * hl;
            pb[prow * 32 + lm]      = (__bf16)e0;
            pb[prow * 32 + 16 + lm] = (__bf16)e1;
        }
        asm volatile("s_wait_dscnt 0" ::: "memory");   // P LDS writes visible to re-read

        // --- P (16x32 A-frag over keys) x V (32x16 B-frags from Vt) ---
        const v16bf pa = ld_a_frag(pb + lm * 32, hl);
        const __bf16* vbase = Vt + (size_t)hb * SEQ + s0 + 16 * hl;
        acc0 = wmma_bf16(pa, ld_b_frag(vbase + (size_t)(0 * 16 + lm) * SEQ), acc0);
        acc1 = wmma_bf16(pa, ld_b_frag(vbase + (size_t)(1 * 16 + lm) * SEQ), acc1);
        acc2 = wmma_bf16(pa, ld_b_frag(vbase + (size_t)(2 * 16 + lm) * SEQ), acc2);
        acc3 = wmma_bf16(pa, ld_b_frag(vbase + (size_t)(3 * 16 + lm) * SEQ), acc3);
    }

#pragma unroll
    for (int r = 0; r < 8; ++r) {
        const float inv = 1.0f / l[r];
        __bf16* orow = O + (size_t)(q0 + r + 8 * hl) * DIMN + hb + lm;
        orow[0]  = (__bf16)(acc0[r] * inv);
        orow[16] = (__bf16)(acc1[r] * inv);
        orow[32] = (__bf16)(acc2[r] * inv);
        orow[48] = (__bf16)(acc3[r] * inv);
    }
}

// ---- host orchestration ----------------------------------------------------

extern "C" void kernel_launch(void* const* d_in, const int* in_sizes, int n_in,
                              void* d_out, int out_size, void* d_ws, size_t ws_size,
                              hipStream_t stream) {
    (void)in_sizes; (void)n_in; (void)out_size; (void)ws_size;
    const float* x  = (const float*)d_in[0];
    const float* wq = (const float*)d_in[1];
    const float* wk = (const float*)d_in[2];
    const float* wv = (const float*)d_in[3];
    const float* wo = (const float*)d_in[4];
    float* out = (float*)d_out;

    char* ws = (char*)d_ws;
    size_t off = 0;
    auto carve = [&](size_t bytes) {
        char* p = ws + off;
        off += (bytes + 255) & ~(size_t)255;
        return p;
    };
    __bf16* xb  = (__bf16*)carve((size_t)SEQ * DIMN * 2);
    __bf16* wqb = (__bf16*)carve((size_t)DIMN * DIMN * 2);
    __bf16* wkb = (__bf16*)carve((size_t)DIMN * DIMN * 2);
    __bf16* wvb = (__bf16*)carve((size_t)DIMN * DIMN * 2);
    __bf16* wob = (__bf16*)carve((size_t)DIMN * DIMN * 2);
    __bf16* qb  = (__bf16*)carve((size_t)SEQ * DIMN * 2);
    __bf16* kb  = (__bf16*)carve((size_t)SEQ * DIMN * 2);
    __bf16* vtb = (__bf16*)carve((size_t)DIMN * SEQ * 2);   // transposed [DIMN, SEQ]
    __bf16* ab  = (__bf16*)carve((size_t)SEQ * DIMN * 2);

    const int nX = SEQ * DIMN;
    const int nW = DIMN * DIMN;
    f32_to_bf16_kernel<<<(nX + 255) / 256, 256, 0, stream>>>(x,  xb,  nX);
    f32_to_bf16_kernel<<<(nW + 255) / 256, 256, 0, stream>>>(wq, wqb, nW);
    f32_to_bf16_kernel<<<(nW + 255) / 256, 256, 0, stream>>>(wk, wkb, nW);
    f32_to_bf16_kernel<<<(nW + 255) / 256, 256, 0, stream>>>(wv, wvb, nW);
    f32_to_bf16_kernel<<<(nW + 255) / 256, 256, 0, stream>>>(wo, wob, nW);

    const int tiles  = (SEQ / 32) * (DIMN / 64);   // 2048 waves, 32x64 per wave
    const int blocks = tiles / 8;                  // 8 waves / block (wave32)
    gemm_bf16_wmma_kernel<0><<<blocks, 256, 0, stream>>>(xb, wqb, qb,  SEQ, DIMN, DIMN);
    gemm_bf16_wmma_kernel<0><<<blocks, 256, 0, stream>>>(xb, wkb, kb,  SEQ, DIMN, DIMN);
    gemm_bf16_wmma_kernel<1><<<blocks, 256, 0, stream>>>(xb, wvb, vtb, SEQ, DIMN, DIMN);

    const int awaves  = NHEADS * (SEQ / 16);       // 4096 waves
    swa_flash_wmma_kernel<<<awaves / 8, 256, 0, stream>>>(qb, kb, vtb, ab);

    gemm_bf16_wmma_kernel<2><<<blocks, 256, 0, stream>>>(ab, wob, out, SEQ, DIMN, DIMN);
}
